// SWSA_36301063585995
// MI455X (gfx1250) — matrix operand order repaired and checked by reading, not compiled, not run on hardware
//
#include <hip/hip_runtime.h>
#include <math.h>

typedef __attribute__((ext_vector_type(16))) __bf16 bf16x16;
typedef __attribute__((ext_vector_type(8)))  __bf16 bf16x8;
typedef __attribute__((ext_vector_type(8)))  float  f32x8;

#define NHEADS 8
#define HD     32
#define CC     256
#define C3     768
#define TT     8
#define HH     64
#define WW     64
#define HW     4096
#define NF     16          // B*T frames
#define NTOK   (NF*HW)     // 65536 tokens

__device__ __forceinline__ bf16x8 ld8(const __bf16* p) { return *(const bf16x8*)p; }
__device__ __forceinline__ bf16x16 cat16(bf16x8 lo, bf16x8 hi) {
    return __builtin_shufflevector(lo, hi, 0,1,2,3,4,5,6,7,8,9,10,11,12,13,14,15);
}
__device__ __forceinline__ f32x8 wmma_bf16(bf16x16 a, bf16x16 b, f32x8 c) {
    return __builtin_amdgcn_wmma_f32_16x16x32_bf16(false, a, false, b, (short)0, c, false, false);
}

// ---------------- pack x (B,T,C,H,W) f32 -> token-major bf16 [tok][ci] --------
__global__ __launch_bounds__(256) void k_pack_x(const float* __restrict__ x,
                                                __bf16* __restrict__ Xb) {
    size_t tid = (size_t)blockIdx.x * 256 + threadIdx.x;   // tid = n*256 + ci
    int ci = (int)(tid & 255);
    size_t n = tid >> 8;
    int hw = (int)(n & (HW - 1));
    int f  = (int)(n >> 12);
    Xb[tid] = (__bf16)x[((size_t)(f * CC + ci)) * HW + hw];
}

// ---------------- pack conv weights (M,256,3,1) f32 -> bf16 [M][dt*256+ci] ----
__global__ __launch_bounds__(256) void k_pack_w(const float* __restrict__ w,
                                                __bf16* __restrict__ out, int M) {
    int tid = blockIdx.x * 256 + threadIdx.x;
    if (tid >= M * C3) return;
    int m = tid / C3, k = tid % C3;
    int dt = k / CC, ci = k % CC;
    out[tid] = (__bf16)w[(size_t)(m * CC + ci) * 3 + dt];
}

// ---------------- attention relative position bias table [h][p][q] f32 -------
__global__ __launch_bounds__(256) void k_bias_attn(const float* __restrict__ rpb,
                                                   float* __restrict__ bias) {
    int tid = blockIdx.x * 256 + threadIdx.x;   // 8*64*64
    int h = tid >> 12, rem = tid & 4095;
    int p = rem >> 6, q = rem & 63;
    int py = p >> 3, px = p & 7, qy = q >> 3, qx = q & 7;
    int idx = (py - qy + 7) * 15 + (px - qx + 7);
    bias[tid] = rpb[idx * NHEADS + h];
}

// ---------------- temporal-conv GEMM --------------------------------------
// D[M, tok] = sum_{dt,ci} W[M][dt*256+ci] * X[tok + (dt-1)*HW][ci]  (+bias)
// mode 0: store bf16 token-major rows of C3 (conv_in).  mode 1: store f32
// planar (frame, co, hw) (conv_out, final output).
__global__ __launch_bounds__(256) void k_gemm(const __bf16* __restrict__ W,
                                              const __bf16* __restrict__ X,
                                              const float* __restrict__ bias,
                                              __bf16* __restrict__ outB,
                                              float* __restrict__ outF,
                                              int Mtot, int mode) {
    const int lane = threadIdx.x & 31;
    const int wid  = threadIdx.x >> 5;
    const int l15  = lane & 15;
    const int hi   = lane >> 4;
    const int nBlock = blockIdx.x * 128;
    const int mBlock = blockIdx.y * 128;
    const int mWave = mBlock + (wid >> 1) * 32;     // 32 rows of M per wave
    const int nWave = nBlock + (wid & 1) * 64;      // 64 tokens per wave
    const int frame = nBlock >> 12;                 // 128 | 4096 -> uniform
    const int t = frame & (TT - 1);

    f32x8 acc[2][4];
    #pragma unroll
    for (int i = 0; i < 2; ++i)
        #pragma unroll
        for (int j = 0; j < 4; ++j) {
            f32x8 z = {0.f,0.f,0.f,0.f,0.f,0.f,0.f,0.f};
            acc[i][j] = z;
        }

    for (int dt = 0; dt < 3; ++dt) {
        int tsrc = t + dt - 1;
        if (tsrc < 0 || tsrc >= TT) continue;
        const __bf16* Xs = X + (size_t)(nWave + (dt - 1) * HW) * CC;
        const __bf16* Wd = W + dt * CC;
        #pragma unroll
        for (int kc = 0; kc < CC; kc += 32) {
            bf16x16 a[2], b[4];
            const int ka = kc + hi * 8;
            #pragma unroll
            for (int i = 0; i < 2; ++i) {
                const __bf16* p = Wd + (size_t)(mWave + i * 16 + l15) * C3 + ka;
                a[i] = cat16(ld8(p), ld8(p + 16));
            }
            const int kb = kc + hi * 8;
            #pragma unroll
            for (int j = 0; j < 4; ++j) {
                const __bf16* p = Xs + (size_t)(j * 16 + l15) * CC + kb;
                b[j] = cat16(ld8(p), ld8(p + 16));
            }
            #pragma unroll
            for (int i = 0; i < 2; ++i)
                #pragma unroll
                for (int j = 0; j < 4; ++j)
                    acc[i][j] = wmma_bf16(a[i], b[j], acc[i][j]);
        }
    }

    #pragma unroll
    for (int i = 0; i < 2; ++i) {
        const int coBase = mWave + i * 16 + hi * 8;
        #pragma unroll
        for (int j = 0; j < 4; ++j) {
            const int n = nWave + j * 16 + l15;
            if (mode == 0) {
                bf16x8 o;
                #pragma unroll
                for (int r = 0; r < 8; ++r)
                    o[r] = (__bf16)(acc[i][j][r] + bias[coBase + r]);
                *(bf16x8*)(outB + (size_t)n * C3 + coBase) = o;
            } else {
                const int hw = n & (HW - 1);
                float* dst = outF + ((size_t)frame * Mtot + coBase) * HW + hw;
                #pragma unroll
                for (int r = 0; r < 8; ++r)
                    dst[(size_t)r * HW] = acc[i][j][r] + bias[coBase + r];
            }
        }
    }
}

// ---------------- depthwise 3x3 per frame, bf16 in/out, f32 accum ---------
__global__ __launch_bounds__(256) void k_dconv(const __bf16* __restrict__ y1,
                                               const float* __restrict__ dw,
                                               const float* __restrict__ db,
                                               __bf16* __restrict__ y2) {
    size_t tid = (size_t)blockIdx.x * 256 + threadIdx.x;   // NF*HW*96 items
    int c8 = (int)(tid % 96);
    size_t pix = tid / 96;
    int hw = (int)(pix & (HW - 1));
    int f  = (int)(pix >> 12);
    int h = hw >> 6, w = hw & 63;
    int c0 = c8 * 8;
    float acc[8] = {0.f,0.f,0.f,0.f,0.f,0.f,0.f,0.f};
    #pragma unroll
    for (int ky = 0; ky < 3; ++ky) {
        int hs = h + ky - 1;
        if (hs < 0 || hs >= HH) continue;
        #pragma unroll
        for (int kx = 0; kx < 3; ++kx) {
            int wsrc = w + kx - 1;
            if (wsrc < 0 || wsrc >= WW) continue;
            bf16x8 v = ld8(y1 + ((size_t)f * HW + hs * WW + wsrc) * C3 + c0);
            #pragma unroll
            for (int r = 0; r < 8; ++r)
                acc[r] += (float)v[r] * dw[(c0 + r) * 9 + ky * 3 + kx];
        }
    }
    bf16x8 o;
    #pragma unroll
    for (int r = 0; r < 8; ++r) o[r] = (__bf16)(acc[r] + db[c0 + r]);
    *(bf16x8*)(y2 + ((size_t)f * HW + hw) * C3 + c0) = o;
}

// ---------------- window attention: one (window, head) per wave -----------
__global__ __launch_bounds__(128) void k_attn(const __bf16* __restrict__ y2,
                                              const float* __restrict__ biasA,
                                              __bf16* __restrict__ ao) {
    __shared__ __bf16 sA[4][64 * 72];   // a matrix [q][key], stride 72
    __shared__ __bf16 sV[4][32 * 72];   // v^T [hd][key], stride 72
    const int lane = threadIdx.x & 31;
    const int wid  = threadIdx.x >> 5;
    const int l15  = lane & 15;
    const int hi   = lane >> 4;
    const int win  = blockIdx.x;                 // 0..1023
    const int head = blockIdx.y * 4 + wid;       // 0..7
    const int f = win >> 6;
    const int wloc = win & 63;
    const int wy = wloc >> 3, wx = wloc & 7;
    const size_t tokBase = (size_t)f * HW + (size_t)wy * 8 * WW + wx * 8;
    const int qoff = head * HD;
    const int koff = CC + head * HD;
    const int voff = 2 * CC + head * HD;

    // stage v^T into LDS: sV[hd][key] = v[key][hd]
    for (int key = 0; key < 64; ++key) {
        size_t tok = tokBase + (size_t)(key >> 3) * WW + (key & 7);
        sV[wid][lane * 72 + key] = y2[tok * C3 + voff + lane];
    }

    // q (A frags) and k^T (B frags), K = 32 = one WMMA step
    bf16x16 qf[4], kf[4];
    #pragma unroll
    for (int i = 0; i < 4; ++i) {
        int qtok = i * 16 + l15;
        size_t tok = tokBase + (size_t)(qtok >> 3) * WW + (qtok & 7);
        const __bf16* pq = y2 + tok * C3 + qoff + hi * 8;
        qf[i] = cat16(ld8(pq), ld8(pq + 16));
        const __bf16* pk = y2 + tok * C3 + koff + hi * 8;
        kf[i] = cat16(ld8(pk), ld8(pk + 16));
    }
    f32x8 att[4][4];
    #pragma unroll
    for (int i = 0; i < 4; ++i)
        #pragma unroll
        for (int j = 0; j < 4; ++j) {
            f32x8 z = {0.f,0.f,0.f,0.f,0.f,0.f,0.f,0.f};
            att[i][j] = wmma_bf16(qf[i], kf[j], z);
        }

    // scale + relative position bias
    const float scale = 0.1767766952966369f;     // 32^-0.5
    #pragma unroll
    for (int i = 0; i < 4; ++i)
        #pragma unroll
        for (int j = 0; j < 4; ++j)
            #pragma unroll
            for (int r = 0; r < 8; ++r) {
                int M = i * 16 + r + hi * 8;
                int N = j * 16 + l15;
                att[i][j][r] = att[i][j][r] * scale +
                               biasA[((size_t)head * 64 + M) * 64 + N];
            }

    // blended activation: 0.25*softmax + 0.75*relu^2 ; write bf16 a to LDS
    #pragma unroll
    for (int i = 0; i < 4; ++i) {
        float mx[8], sm[8];
        #pragma unroll
        for (int r = 0; r < 8; ++r) {
            float m = fmaxf(fmaxf(att[i][0][r], att[i][1][r]),
                            fmaxf(att[i][2][r], att[i][3][r]));
            #pragma unroll
            for (int d = 1; d < 16; d <<= 1) m = fmaxf(m, __shfl_xor(m, d, 32));
            float s = 0.f;
            #pragma unroll
            for (int j = 0; j < 4; ++j) s += __expf(att[i][j][r] - m);
            #pragma unroll
            for (int d = 1; d < 16; d <<= 1) s += __shfl_xor(s, d, 32);
            mx[r] = m; sm[r] = 1.0f / s;
        }
        #pragma unroll
        for (int j = 0; j < 4; ++j) {
            int col = j * 16 + l15;
            #pragma unroll
            for (int r = 0; r < 8; ++r) {
                float x = att[i][j][r];
                float sf = __expf(x - mx[r]) * sm[r];
                float rl = fmaxf(x, 0.f);
                int M = i * 16 + r + hi * 8;
                sA[wid][M * 72 + col] = (__bf16)(0.25f * sf + 0.75f * rl * rl);
            }
        }
    }
    __syncthreads();

    // out = a @ v :  M=64 queries, N=32 hd, K=64 keys (2 WMMA K-steps)
    f32x8 outf[4][2];
    #pragma unroll
    for (int i = 0; i < 4; ++i)
        #pragma unroll
        for (int jn = 0; jn < 2; ++jn) {
            f32x8 z = {0.f,0.f,0.f,0.f,0.f,0.f,0.f,0.f};
            outf[i][jn] = z;
        }
    #pragma unroll
    for (int ks = 0; ks < 64; ks += 32) {
        bf16x16 af[4], vf[2];
        const int ka = ks + hi * 8;
        #pragma unroll
        for (int i = 0; i < 4; ++i) {
            const __bf16* p = &sA[wid][(i * 16 + l15) * 72 + ka];
            af[i] = cat16(ld8(p), ld8(p + 16));
        }
        const int kb = ks + hi * 8;
        #pragma unroll
        for (int jn = 0; jn < 2; ++jn) {
            const __bf16* p = &sV[wid][(jn * 16 + l15) * 72 + kb];
            vf[jn] = cat16(ld8(p), ld8(p + 16));
        }
        #pragma unroll
        for (int i = 0; i < 4; ++i)
            #pragma unroll
            for (int jn = 0; jn < 2; ++jn)
                outf[i][jn] = wmma_bf16(af[i], vf[jn], outf[i][jn]);
    }

    // store token-major bf16 [tok][256]
    #pragma unroll
    for (int i = 0; i < 4; ++i)
        #pragma unroll
        for (int jn = 0; jn < 2; ++jn) {
            int hd = jn * 16 + l15;
            #pragma unroll
            for (int r = 0; r < 8; ++r) {
                int q = i * 16 + r + hi * 8;
                size_t tok = tokBase + (size_t)(q >> 3) * WW + (q & 7);
                ao[tok * CC + head * HD + hd] = (__bf16)outf[i][jn][r];
            }
        }
}

extern "C" void kernel_launch(void* const* d_in, const int* in_sizes, int n_in,
                              void* d_out, int out_size, void* d_ws, size_t ws_size,
                              hipStream_t stream) {
    const float* x          = (const float*)d_in[0];
    const float* conv_in_w  = (const float*)d_in[1];
    const float* conv_in_b  = (const float*)d_in[2];
    const float* dconv_w    = (const float*)d_in[3];
    const float* dconv_b    = (const float*)d_in[4];
    const float* conv_out_w = (const float*)d_in[5];
    const float* conv_out_b = (const float*)d_in[6];
    const float* rpb        = (const float*)d_in[7];

    char* ws = (char*)d_ws;
    __bf16* Xb = (__bf16*)ws;                 ws += (size_t)NTOK * CC * 2;   // 33.5 MB
    __bf16* W1 = (__bf16*)ws;                 ws += (size_t)C3 * C3 * 2;     // 1.18 MB
    __bf16* W2 = (__bf16*)ws;                 ws += (size_t)CC * C3 * 2;     // 0.39 MB
    float*  bA = (float*)ws;                  ws += (size_t)NHEADS * 64 * 64 * 4;
    __bf16* y1 = (__bf16*)ws;                 ws += (size_t)NTOK * C3 * 2;   // 100.7 MB
    __bf16* y2 = (__bf16*)ws;                 ws += (size_t)NTOK * C3 * 2;   // 100.7 MB
    __bf16* ao = (__bf16*)ws;                 ws += (size_t)NTOK * CC * 2;   // 33.5 MB

    k_pack_x   <<<(NTOK * CC) / 256, 256, 0, stream>>>(x, Xb);
    k_pack_w   <<<(C3 * C3) / 256, 256, 0, stream>>>(conv_in_w, W1, C3);
    k_pack_w   <<<(CC * C3) / 256, 256, 0, stream>>>(conv_out_w, W2, CC);
    k_bias_attn<<<(NHEADS * 64 * 64) / 256, 256, 0, stream>>>(rpb, bA);

    // conv_in: 768 x 768 x 65536 GEMM
    k_gemm<<<dim3(NTOK / 128, C3 / 128), 256, 0, stream>>>(
        W1, Xb, conv_in_b, y1, nullptr, C3, 0);

    // depthwise 3x3
    k_dconv<<<(size_t)NF * HW * 96 / 256, 256, 0, stream>>>(y1, dconv_w, dconv_b, y2);

    // window attention: 1024 windows x 8 heads, 4 heads per block
    k_attn<<<dim3(1024, 2), 128, 0, stream>>>(y2, bA, ao);

    // conv_out: 256 x 768 x 65536 GEMM -> f32 planar output
    k_gemm<<<dim3(NTOK / 128, CC / 128), 256, 0, stream>>>(
        W2, ao, conv_out_b, nullptr, (float*)d_out, CC, 1);
}
